// MultiHeadAttention_19361712571420
// MI455X (gfx1250) — matrix operand order repaired and checked by reading, not compiled, not run on hardware
//
#include <hip/hip_runtime.h>
#include <hip/hip_bf16.h>
#include <math.h>

typedef __attribute__((ext_vector_type(16))) _Float16 v16h;
typedef __attribute__((ext_vector_type(8)))  _Float16 v8h;
typedef __attribute__((ext_vector_type(8)))  float    v8f;

#define D_MODEL 1024
#define NUM_HEADS 16
#define DK 64
#define SEQ 2048
#define BATCH 2
#define M_TOK (BATCH * SEQ)   // 4096 tokens

// ---------------- WMMA helpers (CDNA5 16x16x32 f16 -> f32) ----------------

__device__ inline v8f wmma_f16(v16h a, v16h b, v8f c) {
    // 8 args: (neg_a, A, neg_b, B, c_mod, C, reuse_a, reuse_b)
    return __builtin_amdgcn_wmma_f32_16x16x32_f16(false, a, false, b, (short)0, c,
                                                  false, false);
}

__device__ inline void cvt8(v16h& r, int base, const float4& a, const float4& b) {
    r[base + 0] = (_Float16)a.x; r[base + 1] = (_Float16)a.y;
    r[base + 2] = (_Float16)a.z; r[base + 3] = (_Float16)a.w;
    r[base + 4] = (_Float16)b.x; r[base + 5] = (_Float16)b.y;
    r[base + 6] = (_Float16)b.z; r[base + 7] = (_Float16)b.w;
}

// A-matrix 16x32 f16 from f32 row-major source; tile at (row0, k0), leading dim ld.
// Lane l: row m = l&15, half = l>>4. elems 0..7 -> K = half*8+e, elems 8..15 -> K = 16+half*8+(e-8).
__device__ inline v16h load_a_f32(const float* __restrict__ base, int ld, int row0,
                                  int k0, int lane) {
    const int m = lane & 15, hf = lane >> 4;
    const float* p = base + (size_t)(row0 + m) * ld + k0 + hf * 8;
    float4 x0 = *(const float4*)(p);
    float4 x1 = *(const float4*)(p + 4);
    float4 x2 = *(const float4*)(p + 16);
    float4 x3 = *(const float4*)(p + 20);
    v16h a;
    cvt8(a, 0, x0, x1);
    cvt8(a, 8, x2, x3);
    return a;
}

// Same A layout, f16 source (works for global or LDS pointers after inlining).
__device__ inline v16h load_a_f16(const _Float16* base, int ld, int row0,
                                  int k0, int lane) {
    const int m = lane & 15, hf = lane >> 4;
    const _Float16* p = base + (size_t)(row0 + m) * ld + k0 + hf * 8;
    v8h g0 = *(const v8h*)(p);
    v8h g1 = *(const v8h*)(p + 16);
    v16h a;
#pragma unroll
    for (int e = 0; e < 8; ++e) { a[e] = g0[e]; a[e + 8] = g1[e]; }
    return a;
}

// B-matrix 32x16, B[k][n] = src[(col0+n)*ld + k0 + k] (k contiguous in memory).
// Lane l: col n = l&15, half = l>>4. elems 0..7 -> K = half*8+e, elems 8..15 -> K = 16+half*8+(e-8).
__device__ inline v16h load_b_f32(const float* __restrict__ base, int ld, int col0,
                                  int k0, int lane) {
    const int n = lane & 15, hf = lane >> 4;
    const float* p = base + (size_t)(col0 + n) * ld + k0 + hf * 8;
    float4 x0 = *(const float4*)(p);
    float4 x1 = *(const float4*)(p + 4);
    float4 x2 = *(const float4*)(p + 16);
    float4 x3 = *(const float4*)(p + 20);
    v16h b;
    cvt8(b, 0, x0, x1);
    cvt8(b, 8, x2, x3);
    return b;
}

__device__ inline v16h load_b_f16(const _Float16* __restrict__ base, int ld, int col0,
                                  int k0, int lane) {
    const int n = lane & 15, hf = lane >> 4;
    const _Float16* p = base + (size_t)(col0 + n) * ld + k0 + hf * 8;
    v8h g0 = *(const v8h*)(p);
    v8h g1 = *(const v8h*)(p + 16);
    v16h b;
#pragma unroll
    for (int e = 0; e < 8; ++e) { b[e] = g0[e]; b[e + 8] = g1[e]; }
    return b;
}

// ---------------- Stage 1: fused Q/K/V projections ----------------
// q,k -> f16 [B,H,L,dk]; v -> f16 [B,H,dk,L] (transposed so AV B-loads are contiguous)
__global__ void qkv_proj_kernel(const float* __restrict__ Q, const float* __restrict__ Kin,
                                const float* __restrict__ V,
                                const float* __restrict__ Wq, const float* __restrict__ bq,
                                const float* __restrict__ Wk, const float* __restrict__ bk,
                                const float* __restrict__ Wv, const float* __restrict__ bv,
                                _Float16* __restrict__ qh, _Float16* __restrict__ kh,
                                _Float16* __restrict__ vh) {
    const float* X; const float* W; const float* bias; _Float16* dst;
    if (blockIdx.z == 0)      { X = Q;   W = Wq; bias = bq; dst = qh; }
    else if (blockIdx.z == 1) { X = Kin; W = Wk; bias = bk; dst = kh; }
    else                      { X = V;   W = Wv; bias = bv; dst = vh; }

    const int lane = threadIdx.x & 31;
    const int wave = threadIdx.x >> 5;                 // 8 waves: 2 (M) x 4 (N)
    const int row0 = blockIdx.x * 32 + (wave & 1) * 16;
    const int col0 = blockIdx.y * 64 + (wave >> 1) * 16;

    v8f c = {};
    for (int k0 = 0; k0 < D_MODEL; k0 += 32) {
        v16h a = load_a_f32(X, D_MODEL, row0, k0, lane);
        v16h b = load_b_f32(W, D_MODEL, col0, k0, lane);
        c = wmma_f16(a, b, c);
    }

    const int n = lane & 15, hf = lane >> 4;
    const int o = col0 + n;
    const float bias_o = bias[o];
    const int h = o >> 6, d = o & 63;

    _Float16 vals[8];
#pragma unroll
    for (int r = 0; r < 8; ++r) vals[r] = (_Float16)(c[r] + bias_o);

    if (blockIdx.z == 2) {
        // v stored transposed: [B,H,dk,L]
#pragma unroll
        for (int r = 0; r < 8; ++r) {
            const int tok = row0 + r + hf * 8;
            const int bb = tok >> 11, l = tok & (SEQ - 1);
            dst[(((size_t)(bb * NUM_HEADS + h)) * DK + d) * SEQ + l] = vals[r];
        }
    } else {
#pragma unroll
        for (int r = 0; r < 8; ++r) {
            const int tok = row0 + r + hf * 8;
            const int bb = tok >> 11, l = tok & (SEQ - 1);
            dst[(((size_t)(bb * NUM_HEADS + h)) * SEQ + l) * DK + d] = vals[r];
        }
    }
}

// ---------------- Stage 2: scores + softmax + AV, one block per (b,h,16 q-rows) ----------
__global__ void attention_kernel(const _Float16* __restrict__ qh,
                                 const _Float16* __restrict__ kh,
                                 const _Float16* __restrict__ vh,
                                 float* __restrict__ attn, _Float16* __restrict__ ctx) {
    __shared__ float    S[16 * SEQ];                   // 128 KB: raw scores / reduce scratch
    __shared__ _Float16 Pf[16 * SEQ];                  // 64 KB: normalized probs, f16

    const int qb = blockIdx.x, h = blockIdx.y, b = blockIdx.z;
    const int lane = threadIdx.x & 31, wave = threadIdx.x >> 5;
    const int n = lane & 15, hf = lane >> 4;
    const size_t headoff = (size_t)(b * NUM_HEADS + h) * SEQ * DK;
    const _Float16* qhead  = qh + headoff;
    const _Float16* khead  = kh + headoff;
    const _Float16* vThead = vh + headoff;             // [dk][SEQ]
    const int q0 = qb * 16;

    // ---- warm L2 with the V head (256 KB) while scores+softmax run.
    // 256 threads x 8 x 128B lines = full head; lowers to global_prefetch_b8.
    {
        const _Float16* pv = vThead + (size_t)threadIdx.x * 512;
#pragma unroll
        for (int i = 0; i < 8; ++i)
            __builtin_prefetch(pv + i * 64, 0, 1);
    }

    // ---- scores: S = (q_tile @ K^T) * 1/sqrt(dk); wave w covers keys [w*256, w*256+256)
    {
        v16h a0 = load_a_f16(qhead, DK, q0, 0, lane);
        v16h a1 = load_a_f16(qhead, DK, q0, 32, lane);
        for (int t = 0; t < 16; ++t) {
            const int kc0 = (wave * 16 + t) * 16;
            v8f c = {};
            c = wmma_f16(a0, load_b_f16(khead, DK, kc0, 0, lane), c);
            c = wmma_f16(a1, load_b_f16(khead, DK, kc0, 32, lane), c);
#pragma unroll
            for (int r = 0; r < 8; ++r)
                S[(r + hf * 8) * SEQ + kc0 + n] = c[r] * 0.125f;
        }
    }
    __syncthreads();

    // ---- softmax over each of the 16 rows; 16 consecutive lanes per row.
    // Writes attn (f32, non-temporal streaming store) and Pf (f16, for AV).
    {
        const int row = threadIdx.x >> 4, c16 = threadIdx.x & 15;
        float* Sr = S + row * SEQ;
        float mx = -3.4e38f;
        for (int c = c16; c < SEQ; c += 16) mx = fmaxf(mx, Sr[c]);
#pragma unroll
        for (int off = 8; off > 0; off >>= 1) mx = fmaxf(mx, __shfl_xor(mx, off, 16));
        float sum = 0.f;
        for (int c = c16; c < SEQ; c += 16) {
            float e = __expf(Sr[c] - mx);
            Sr[c] = e;
            sum += e;
        }
#pragma unroll
        for (int off = 8; off > 0; off >>= 1) sum += __shfl_xor(sum, off, 16);
        const float inv = 1.0f / sum;
        float* arow = attn + (((size_t)(b * NUM_HEADS + h)) * SEQ + q0 + row) * SEQ;
        _Float16* prow = Pf + row * SEQ;
        for (int c = c16; c < SEQ; c += 16) {
            float p = Sr[c] * inv;
            __builtin_nontemporal_store(p, &arow[c]);  // 536 MB stream: keep out of L2
            prow[c] = (_Float16)p;
        }
    }
    __syncthreads();

    // ---- AV: out_tile(16x64) = P(16x2048) @ V(2048x64), V transposed in memory.
    // 8 waves: n-tile = wave&3, K-half = wave>>2; reduce pairs through S.
    const int n0 = (wave & 3) * 16;
    const int kbase = (wave >> 2) * (SEQ / 2);
    v8f c = {};
    for (int kk = 0; kk < SEQ / 2; kk += 32) {
        const int k0 = kbase + kk;
        v16h a  = load_a_f16(Pf, SEQ, 0, k0, lane);              // from LDS
        v16h bb = load_b_f16(vThead, SEQ, n0, k0, lane);         // contiguous per lane
        c = wmma_f16(a, bb, c);
    }
    if (wave >= 4) {
#pragma unroll
        for (int r = 0; r < 8; ++r)
            S[(wave & 3) * 256 + (r + hf * 8) * 16 + n] = c[r];
    }
    __syncthreads();
    if (wave < 4) {
#pragma unroll
        for (int r = 0; r < 8; ++r) {
            const float v2 = S[wave * 256 + (r + hf * 8) * 16 + n];
            const int tok = q0 + r + hf * 8;
            ctx[((size_t)b * SEQ + tok) * D_MODEL + h * DK + n0 + n] =
                (_Float16)(c[r] + v2);
        }
    }
}

// ---------------- Stage 3: out = ctx @ Wo^T + bo (f32 output) ----------------
__global__ void out_proj_kernel(const _Float16* __restrict__ ctx,
                                const float* __restrict__ Wo, const float* __restrict__ bo,
                                float* __restrict__ out) {
    const int lane = threadIdx.x & 31;
    const int wave = threadIdx.x >> 5;
    const int row0 = blockIdx.x * 32 + (wave & 1) * 16;
    const int col0 = blockIdx.y * 64 + (wave >> 1) * 16;

    v8f c = {};
    for (int k0 = 0; k0 < D_MODEL; k0 += 32) {
        v16h a = load_a_f16(ctx, D_MODEL, row0, k0, lane);
        v16h b = load_b_f32(Wo, D_MODEL, col0, k0, lane);
        c = wmma_f16(a, b, c);
    }

    const int n = lane & 15, hf = lane >> 4;
    const float bias_o = bo[col0 + n];
#pragma unroll
    for (int r = 0; r < 8; ++r) {
        const int tok = row0 + r + hf * 8;
        out[(size_t)tok * D_MODEL + col0 + n] = c[r] + bias_o;
    }
}

// ---------------- host launcher ----------------
extern "C" void kernel_launch(void* const* d_in, const int* in_sizes, int n_in,
                              void* d_out, int out_size, void* d_ws, size_t ws_size,
                              hipStream_t stream) {
    const float* Q  = (const float*)d_in[0];
    const float* Kx = (const float*)d_in[1];
    const float* V  = (const float*)d_in[2];
    const float* Wq = (const float*)d_in[3];
    const float* bq = (const float*)d_in[4];
    const float* Wk = (const float*)d_in[5];
    const float* bk = (const float*)d_in[6];
    const float* Wv = (const float*)d_in[7];
    const float* bv = (const float*)d_in[8];
    const float* Wo = (const float*)d_in[9];
    const float* bo = (const float*)d_in[10];

    float* out  = (float*)d_out;
    float* attn = out + (size_t)BATCH * SEQ * D_MODEL;   // tuple order: (out, attn)

    const size_t per_head_tensor = (size_t)BATCH * NUM_HEADS * SEQ * DK;  // 4.19M f16
    _Float16* qh  = (_Float16*)d_ws;
    _Float16* kh  = qh + per_head_tensor;
    _Float16* vh  = kh + per_head_tensor;
    _Float16* ctx = vh + per_head_tensor;               // total 32 MB of workspace

    qkv_proj_kernel<<<dim3(M_TOK / 32, D_MODEL / 64, 3), 256, 0, stream>>>(
        Q, Kx, V, Wq, bq, Wk, bk, Wv, bv, qh, kh, vh);
    attention_kernel<<<dim3(SEQ / 16, NUM_HEADS, BATCH), 256, 0, stream>>>(
        qh, kh, vh, attn, ctx);
    out_proj_kernel<<<dim3(M_TOK / 32, D_MODEL / 64, 1), 256, 0, stream>>>(
        ctx, Wo, bo, out);
}
